// Dec_36713380446238
// MI455X (gfx1250) — matrix operand order repaired and checked by reading, not compiled, not run on hardware
//
#include <hip/hip_runtime.h>
#include <hip/hip_bf16.h>

// ---------------------------------------------------------------------------
// Bidirectional 2-layer SRU for MI455X (gfx1250, wave32, WMMA).
// Memory-bound: bf16 storage + v_wmma_f32_16x16x32_bf16 + f32 accumulate.
// A-tiles staged with GLOBAL_LOAD_ASYNC_TO_LDS_B128 (ASYNCcnt, double buffer).
// ---------------------------------------------------------------------------

typedef float  v8f   __attribute__((ext_vector_type(8)));
typedef __bf16 v16bf __attribute__((ext_vector_type(16)));
typedef __bf16 v8bf  __attribute__((ext_vector_type(8)));

static __device__ __forceinline__ __bf16 f2bf(float f) { return (__bf16)f; }
static __device__ __forceinline__ float  bf2f(__bf16 h) { return (float)h; }
static __device__ __forceinline__ float  sigmoidf_(float x) {
  return 1.0f / (1.0f + __expf(-x));
}

// CDNA5 async global->LDS copy (16 B per lane), tracked by ASYNCcnt.
// LDS dest address = low 32 bits of generic pointer (flat LDS aperture keeps
// the LDS byte offset in addr[31:0], per CDNA5 ISA).
static __device__ __forceinline__ void async_copy_b128(void* lds_ptr,
                                                       const void* gptr) {
  unsigned int lds_addr = (unsigned int)(unsigned long long)lds_ptr;
  asm volatile("global_load_async_to_lds_b128 %0, %1, off"
               :
               : "v"(lds_addr), "v"(gptr)
               : "memory");
}
static __device__ __forceinline__ void wait_async0() {
#if __has_builtin(__builtin_amdgcn_s_wait_asynccnt)
  __builtin_amdgcn_s_wait_asynccnt(0);
#else
  asm volatile("s_wait_asynccnt 0x0" ::: "memory");
#endif
}

// Problem constants
#define Tt   2048
#define Bb   32
#define TBc  (Tt * Bb)            // 65536 rows

// ---------------------------------------------------------------------------
// Pack f32 weight [K,N] (row-major) into per-lane WMMA B-fragment order:
// layout [K/32][N/16][lane(32)][elem(16)] bf16.
// Contiguous K order used here (A matches): B 32x16 -> lanes 0-15: col N=lane, K=0..15
// sequential; lanes 16-31: col N=lane-16, K=16..31 sequential.
// ---------------------------------------------------------------------------
__global__ __launch_bounds__(256) void pack_b_kernel(
    const float* __restrict__ W, __bf16* __restrict__ Bp, int K, int N) {
  int idx = blockIdx.x * 256 + threadIdx.x;
  if (idx >= K * N) return;
  int e     = idx & 15;          // element within lane
  int lane  = (idx >> 4) & 31;   // lane
  int blk   = idx >> 9;          // which 512-element fragment block
  int nb16  = N >> 4;
  int n16   = blk % nb16;
  int kb    = blk / nb16;
  int n     = n16 * 16 + (lane & 15);
  int k     = kb * 32 + (lane >> 4) * 16 + e;
  Bp[idx] = f2bf(W[(size_t)k * N + n]);
}

// ---------------------------------------------------------------------------
// LayerNorm: one wave32 per row, shfl_xor reductions, bf16 output.
// Templated input type (f32 for layer 0 input, bf16 for h1).
// ---------------------------------------------------------------------------
template <int ROW, typename XT>
__global__ __launch_bounds__(256) void ln_rows_kernel(
    const XT* __restrict__ X, const float* __restrict__ g,
    const float* __restrict__ bta, __bf16* __restrict__ Y, int nRows) {
  constexpr int PER = ROW / 32;
  int row  = blockIdx.x * 8 + (threadIdx.x >> 5);
  int lane = threadIdx.x & 31;
  if (row >= nRows) return;
  const XT* xr = X + (size_t)row * ROW;
  float v[PER];
  float s = 0.f;
#pragma unroll
  for (int i = 0; i < PER; ++i) { v[i] = (float)xr[i * 32 + lane]; s += v[i]; }
#pragma unroll
  for (int o = 16; o > 0; o >>= 1) s += __shfl_xor(s, o, 32);
  float mean = s * (1.0f / ROW);
  float q = 0.f;
#pragma unroll
  for (int i = 0; i < PER; ++i) { float d = v[i] - mean; q += d * d; }
#pragma unroll
  for (int o = 16; o > 0; o >>= 1) q += __shfl_xor(q, o, 32);
  float rstd = rsqrtf(q * (1.0f / ROW) + 1e-5f);
  __bf16* yr = Y + (size_t)row * ROW;
#pragma unroll
  for (int i = 0; i < PER; ++i) {
    int col = i * 32 + lane;
    yr[col] = f2bf((v[i] - mean) * rstd * g[col] + bta[col]);
  }
}

// ---------------------------------------------------------------------------
// WMMA GEMM: C[M,N](bf16) = A[M,K](bf16, row-major) * Bp(packed fragments).
// Block = 256 threads = 8 waves; tile BM=128 x BN=64, BK=32.
// Wave grid 4(M) x 2(N); each wave computes 2x2 16x16 WMMA tiles.
// A double-buffered in LDS via async copies; padded stride 40 halfwords
// (80 B) -> conflict-free 16-lane row reads, 16 B aligned for ds_read_b128.
// One barrier per K-step: async writes for tile k+1 target the buffer last
// read at k-1, and all waves passed this iteration's barrier (issued after
// k-1's compute) before the writes are issued.
// ---------------------------------------------------------------------------
#define BMt 128
#define BNt 64
#define BKt 32
#define LDA 40

__global__ __launch_bounds__(256) void gemm_bf16_kernel(
    const __bf16* __restrict__ A, const __bf16* __restrict__ Bp,
    __bf16* __restrict__ C, int M, int N, int K) {
  __shared__ __align__(16) __bf16 sA[2][BMt * LDA];

  const int tid    = threadIdx.x;
  const int lane   = tid & 31;
  const int wid    = tid >> 5;
  const int waveN  = wid & 1;    // 0..1
  const int waveM  = wid >> 1;   // 0..3
  const int mBlock = blockIdx.y * BMt;
  const int nBlock = blockIdx.x * BNt;
  const int nb16   = N >> 4;
  const int lrow   = lane & 15;
  const int khalf  = lane >> 4;

  // Staging assignment: 512 x 16B chunks; this thread owns rows tid>>2 and
  // (tid>>2)+64, column segment (tid&3)*8 halfwords.
  const int segs = (tid & 3) * 8;
  const int rowA = tid >> 2;
  const int rowB = rowA + 64;

  v8f acc[2][2] = {};

  const int kIters = K >> 5;

  auto issue_tile = [&](int kb, int buf) {
    const __bf16* g0 = A + (size_t)(mBlock + rowA) * K + kb * BKt + segs;
    const __bf16* g1 = A + (size_t)(mBlock + rowB) * K + kb * BKt + segs;
    async_copy_b128(&sA[buf][rowA * LDA + segs], g0);
    async_copy_b128(&sA[buf][rowB * LDA + segs], g1);
  };

  issue_tile(0, 0);

  for (int kb = 0; kb < kIters; ++kb) {
    const int cur = kb & 1;
    wait_async0();       // this wave's copies for sA[cur] landed
    __syncthreads();     // everyone's copies landed; k-1 compute finished
    if (kb + 1 < kIters) issue_tile(kb + 1, cur ^ 1);
    if (kb + 2 < kIters) {
      __builtin_prefetch(A + (size_t)(mBlock + rowA) * K + (kb + 2) * BKt, 0,
                         1);
    }

    // B fragments: contiguous 32 B per lane from the packed weight buffer.
    v16bf bfrag[2];
#pragma unroll
    for (int j = 0; j < 2; ++j) {
      int n16 = (nBlock + waveN * 32 + j * 16) >> 4;
      bfrag[j] =
          *(const v16bf*)(Bp + (((size_t)kb * nb16 + n16) * 32 + lane) * 16);
    }

    // A fragments in contiguous K order (matching packed B):
    // lanes 0-15: K 0..15 ; lanes 16-31: K 16..31.
    const __bf16* sAc = &sA[cur][0];
#pragma unroll
    for (int i = 0; i < 2; ++i) {
      int row = waveM * 32 + i * 16 + lrow;
      const __bf16* ra = sAc + row * LDA + khalf * 16;
      v8bf lo = *(const v8bf*)(ra);
      v8bf hi = *(const v8bf*)(ra + 8);
      v16bf afrag = __builtin_shufflevector(lo, hi, 0, 1, 2, 3, 4, 5, 6, 7, 8,
                                            9, 10, 11, 12, 13, 14, 15);
#pragma unroll
      for (int j = 0; j < 2; ++j) {
        acc[i][j] = __builtin_amdgcn_wmma_f32_16x16x32_bf16(
            false, afrag, false, bfrag[j], (short)0, acc[i][j], false, false);
      }
    }
  }

  // C/D layout: VGPR v -> M = v (lanes 0-15) / v+8 (lanes 16-31), N = lane&15.
#pragma unroll
  for (int i = 0; i < 2; ++i) {
    int m0 = mBlock + waveM * 32 + i * 16 + khalf * 8;
#pragma unroll
    for (int j = 0; j < 2; ++j) {
      int n = nBlock + waveN * 32 + j * 16 + lrow;
#pragma unroll
      for (int vi = 0; vi < 8; ++vi) {
        C[(size_t)(m0 + vi) * N + n] = f2bf(acc[i][j][vi]);
      }
    }
  }
}

// ---------------------------------------------------------------------------
// Layer-0 scan (both directions). One thread per (dir,b,h); 2048 sequential
// steps of c = f*c + (1-f)*xt ; h1 = r*c + (1-r)*x0 (identity highway).
// Coalesced bf16 reads of U (h-contiguous across threads).
// ---------------------------------------------------------------------------
__global__ __launch_bounds__(256) void scan0_kernel(
    const __bf16* __restrict__ U0f, const __bf16* __restrict__ U0b,
    const __bf16* __restrict__ x0, const float* __restrict__ c0,
    const float* __restrict__ bf0f, const float* __restrict__ br0f,
    const float* __restrict__ bf0b, const float* __restrict__ br0b,
    __bf16* __restrict__ h1) {
  int idx = blockIdx.x * 256 + threadIdx.x;  // 0..16383
  int dir = idx >> 13;                       // uniform per block
  int bh  = idx & 8191;
  int b   = bh >> 8;
  int h   = bh & 255;
  const __bf16* U = dir ? U0b : U0f;
  float bfv = dir ? bf0b[h] : bf0f[h];
  float brv = dir ? br0b[h] : br0f[h];
  float c   = c0[dir * 8192 + bh];
#pragma unroll 4
  for (int s = 0; s < Tt; ++s) {
    int t = dir ? (Tt - 1 - s) : s;
    size_t row   = (size_t)t * Bb + b;
    size_t ubase = row * 768;
    float xt = bf2f(U[ubase + h]);
    float f  = sigmoidf_(bf2f(U[ubase + 256 + h]) + bfv);
    float r  = sigmoidf_(bf2f(U[ubase + 512 + h]) + brv);
    c = f * c + (1.0f - f) * xt;
    float skip = bf2f(x0[row * 256 + h]);
    h1[row * 512 + dir * 256 + h] = f2bf(r * c + (1.0f - r) * skip);
  }
}

// ---------------------------------------------------------------------------
// Layer-1 scan: only h2[T-1] is needed downstream. Forward direction runs the
// full recurrence for c; reverse direction at t=T-1 is a single step.
// Skip path is the projected highway S1 = x1 @ Wh.
// ---------------------------------------------------------------------------
__global__ __launch_bounds__(256) void scan1_kernel(
    const __bf16* __restrict__ U1f, const __bf16* __restrict__ U1b,
    const __bf16* __restrict__ S1f, const __bf16* __restrict__ S1b,
    const float* __restrict__ c0, const float* __restrict__ bf1f,
    const float* __restrict__ br1f, const float* __restrict__ bf1b,
    const float* __restrict__ br1b, float* __restrict__ h2last) {
  int idx = blockIdx.x * 256 + threadIdx.x;
  int dir = idx >> 13;
  int bh  = idx & 8191;
  int b   = bh >> 8;
  int h   = bh & 255;
  if (dir == 0) {
    float bfv = bf1f[h];
    float c   = c0[2 * 8192 + bh];
#pragma unroll 4
    for (int t = 0; t < Tt; ++t) {
      size_t ubase = ((size_t)t * Bb + b) * 768;
      float xt = bf2f(U1f[ubase + h]);
      float f  = sigmoidf_(bf2f(U1f[ubase + 256 + h]) + bfv);
      c = f * c + (1.0f - f) * xt;
    }
    size_t row = (size_t)(Tt - 1) * Bb + b;
    float r    = sigmoidf_(bf2f(U1f[row * 768 + 512 + h]) + br1f[h]);
    float skip = bf2f(S1f[row * 256 + h]);
    h2last[b * 512 + h] = r * c + (1.0f - r) * skip;
  } else {
    size_t row   = (size_t)(Tt - 1) * Bb + b;
    size_t ubase = row * 768;
    float c  = c0[3 * 8192 + bh];
    float xt = bf2f(U1b[ubase + h]);
    float f  = sigmoidf_(bf2f(U1b[ubase + 256 + h]) + bf1b[h]);
    float r  = sigmoidf_(bf2f(U1b[ubase + 512 + h]) + br1b[h]);
    c = f * c + (1.0f - f) * xt;
    float skip = bf2f(S1b[row * 256 + h]);
    h2last[b * 512 + 256 + h] = r * c + (1.0f - r) * skip;
  }
}

// ---------------------------------------------------------------------------
// Final FC: [32,512] @ [512,2] + bias. 64 dot products; trivial.
// ---------------------------------------------------------------------------
__global__ __launch_bounds__(64) void fc_kernel(
    const float* __restrict__ h2, const float* __restrict__ Wfc,
    const float* __restrict__ bfc, float* __restrict__ out) {
  int tid = threadIdx.x;
  if (tid >= 64) return;
  int b = tid >> 1, o = tid & 1;
  float s = bfc[o];
  for (int k = 0; k < 512; ++k) s = fmaf(h2[b * 512 + k], Wfc[k * 2 + o], s);
  out[b * 2 + o] = s;
}

// ---------------------------------------------------------------------------
// Host orchestration
// ---------------------------------------------------------------------------
static inline size_t align_up(size_t v, size_t a) { return (v + a - 1) & ~(a - 1); }

extern "C" void kernel_launch(void* const* d_in, const int* in_sizes, int n_in,
                              void* d_out, int out_size, void* d_ws,
                              size_t ws_size, hipStream_t stream) {
  const float* x    = (const float*)d_in[0];
  const float* c0   = (const float*)d_in[1];
  const float* ln0g = (const float*)d_in[2];
  const float* ln0b = (const float*)d_in[3];
  const float* ln1g = (const float*)d_in[4];
  const float* ln1b = (const float*)d_in[5];
  const float* W0f  = (const float*)d_in[6];
  const float* bf0f = (const float*)d_in[7];
  const float* br0f = (const float*)d_in[8];
  const float* W0b  = (const float*)d_in[9];
  const float* bf0b = (const float*)d_in[10];
  const float* br0b = (const float*)d_in[11];
  const float* W1f  = (const float*)d_in[12];
  const float* Wh1f = (const float*)d_in[13];
  const float* bf1f = (const float*)d_in[14];
  const float* br1f = (const float*)d_in[15];
  const float* W1b  = (const float*)d_in[16];
  const float* Wh1b = (const float*)d_in[17];
  const float* bf1b = (const float*)d_in[18];
  const float* br1b = (const float*)d_in[19];
  const float* Wfc  = (const float*)d_in[20];
  const float* bfc  = (const float*)d_in[21];
  float* out = (float*)d_out;

  // Workspace carving (bf16 intermediates to halve HBM traffic)
  char* p = (char*)d_ws;
  auto carve = [&](size_t bytes) {
    void* r = (void*)p;
    p += align_up(bytes, 256);
    return r;
  };
  __bf16* x0     = (__bf16*)carve((size_t)TBc * 256 * 2);
  __bf16* x1     = (__bf16*)carve((size_t)TBc * 512 * 2);
  __bf16* U0f    = (__bf16*)carve((size_t)TBc * 768 * 2);
  __bf16* U0b    = (__bf16*)carve((size_t)TBc * 768 * 2);
  __bf16* U1f    = (__bf16*)carve((size_t)TBc * 768 * 2);
  __bf16* U1b    = (__bf16*)carve((size_t)TBc * 768 * 2);
  __bf16* S1f    = (__bf16*)carve((size_t)TBc * 256 * 2);
  __bf16* S1b    = (__bf16*)carve((size_t)TBc * 256 * 2);
  __bf16* h1     = (__bf16*)carve((size_t)TBc * 512 * 2);
  float*  h2last = (float*)carve((size_t)32 * 512 * 4);
  __bf16* W0f_p  = (__bf16*)carve((size_t)256 * 768 * 2);
  __bf16* W0b_p  = (__bf16*)carve((size_t)256 * 768 * 2);
  __bf16* W1f_p  = (__bf16*)carve((size_t)512 * 768 * 2);
  __bf16* W1b_p  = (__bf16*)carve((size_t)512 * 768 * 2);
  __bf16* Wh1f_p = (__bf16*)carve((size_t)512 * 256 * 2);
  __bf16* Wh1b_p = (__bf16*)carve((size_t)512 * 256 * 2);

  // 1) Pack weights into WMMA B-fragment order (tiny, L2-resident)
  pack_b_kernel<<<(256 * 768 + 255) / 256, 256, 0, stream>>>(W0f, W0f_p, 256, 768);
  pack_b_kernel<<<(256 * 768 + 255) / 256, 256, 0, stream>>>(W0b, W0b_p, 256, 768);
  pack_b_kernel<<<(512 * 768 + 255) / 256, 256, 0, stream>>>(W1f, W1f_p, 512, 768);
  pack_b_kernel<<<(512 * 768 + 255) / 256, 256, 0, stream>>>(W1b, W1b_p, 512, 768);
  pack_b_kernel<<<(512 * 256 + 255) / 256, 256, 0, stream>>>(Wh1f, Wh1f_p, 512, 256);
  pack_b_kernel<<<(512 * 256 + 255) / 256, 256, 0, stream>>>(Wh1b, Wh1b_p, 512, 256);

  // 2) LN0 -> x0 (bf16)
  ln_rows_kernel<256, float><<<TBc / 8, 256, 0, stream>>>(x, ln0g, ln0b, x0, TBc);

  // 3) Layer-0 GEMMs: U = x0 @ W  (M=65536, N=768, K=256)
  gemm_bf16_kernel<<<dim3(768 / BNt, TBc / BMt), 256, 0, stream>>>(
      x0, W0f_p, U0f, TBc, 768, 256);
  gemm_bf16_kernel<<<dim3(768 / BNt, TBc / BMt), 256, 0, stream>>>(
      x0, W0b_p, U0b, TBc, 768, 256);

  // 4) Layer-0 scans -> h1 (bf16)
  scan0_kernel<<<64, 256, 0, stream>>>(U0f, U0b, x0, c0, bf0f, br0f, bf0b,
                                       br0b, h1);

  // 5) LN1 -> x1 (bf16)
  ln_rows_kernel<512, __bf16><<<TBc / 8, 256, 0, stream>>>(h1, ln1g, ln1b, x1,
                                                           TBc);

  // 6) Layer-1 GEMMs: U1 (K=512,N=768) and projected highways S1 (K=512,N=256)
  gemm_bf16_kernel<<<dim3(768 / BNt, TBc / BMt), 256, 0, stream>>>(
      x1, W1f_p, U1f, TBc, 768, 512);
  gemm_bf16_kernel<<<dim3(768 / BNt, TBc / BMt), 256, 0, stream>>>(
      x1, W1b_p, U1b, TBc, 768, 512);
  gemm_bf16_kernel<<<dim3(256 / BNt, TBc / BMt), 256, 0, stream>>>(
      x1, Wh1f_p, S1f, TBc, 256, 512);
  gemm_bf16_kernel<<<dim3(256 / BNt, TBc / BMt), 256, 0, stream>>>(
      x1, Wh1b_p, S1b, TBc, 256, 512);

  // 7) Layer-1 scan -> h2[T-1] only
  scan1_kernel<<<64, 256, 0, stream>>>(U1f, U1b, S1f, S1b, c0, bf1f, br1f,
                                       bf1b, br1b, h2last);

  // 8) Final FC -> out [32,2]
  fc_kernel<<<1, 64, 0, stream>>>(h2last, Wfc, bfc, out);
}
